// AprilMultiHeadAttention_55594056680171
// MI455X (gfx1250) — compile-verified, matrix operand order checked by reading, not run on hardware
//
#include <hip/hip_runtime.h>
#include <hip/hip_bf16.h>
#include <math.h>

// ---- problem constants ----
constexpr int Hh = 12;    // heads
constexpr int Bb = 4;     // batch
constexpr int Tt = 1024;  // seq len
constexpr int Ee = 768;   // embed
constexpr int Dd = 768;   // per-head out dim

typedef __bf16 v16bf __attribute__((ext_vector_type(16)));
typedef float  v8f   __attribute__((ext_vector_type(8)));

union FragBF {
  v16bf v;
  unsigned int d[8];
};

// CDNA5 async global->LDS path (ASYNCcnt); falls back to sync LDS stores when
// the builtins are absent (e.g. host pass).
#if defined(__AMDGCN__) && \
    __has_builtin(__builtin_amdgcn_global_load_async_to_lds_b128) && \
    __has_builtin(__builtin_amdgcn_s_wait_asynccnt)
#define USE_ASYNC_LDS 1
#else
#define USE_ASYNC_LDS 0
#endif

#if USE_ASYNC_LDS
// Matches the builtin's parameter type:
//   '__attribute__((__vector_size__(4 * sizeof(int)))) int __device__ *'  (AS1)
// and the LDS side in AS3.
typedef int vi4 __attribute__((vector_size(4 * sizeof(int))));
typedef __attribute__((address_space(1))) vi4* gvi4_p;
typedef __attribute__((address_space(3))) vi4* lvi4_p;
#endif

__device__ __forceinline__ void cp16(unsigned short* l, const unsigned short* g) {
#if USE_ASYNC_LDS
  __builtin_amdgcn_global_load_async_to_lds_b128(
      (gvi4_p)(unsigned short*)g, (lvi4_p)l, 0, 0);
#else
  *(int4*)l = *(const int4*)g;
#endif
}

__device__ __forceinline__ void wait_async() {
#if USE_ASYNC_LDS
  __builtin_amdgcn_s_wait_asynccnt(0);
#endif
}

__device__ __forceinline__ unsigned short f32_to_bf16(float f) {
  unsigned int u = __float_as_uint(f);
  u += 0x7FFFu + ((u >> 16) & 1u);   // round-to-nearest-even
  return (unsigned short)(u >> 16);
}
__device__ __forceinline__ float bf16_to_f32(unsigned short s) {
  return __uint_as_float(((unsigned int)s) << 16);
}

// K-index map for the CDNA5 16-bit A-matrix VGPR layout:
// VGPR v (pair of bf16), lanes 0-15 hold K={0..7,16..23}, lanes 16-31 hold K+8.
__device__ __forceinline__ int kmap(int v, int half) {
  return ((v < 4) ? (2 * v) : (16 + 2 * (v - 4))) + (half ? 8 : 0);
}

// ---- LDS tile geometry ----
// A-style tile: 128 rows x 32 bf16, row stride 40 ushorts (80B: 16B-aligned,
// and 20-uint stride -> 16 distinct banks across the 16 fragment lanes).
constexpr int ASTR_H = 40;                 // ushorts
constexpr int ASTR_U = 20;                 // uints
constexpr int ATILE_H = 128 * ASTR_H;      // per buffer
// B tile: 32 rows (k) x 128 bf16 (n), row stride 136 ushorts (272B, 16B-aligned)
constexpr int BSTR_H = 136;
constexpr int BTILE_H = 32 * BSTR_H;

// A-style 16x32 bf16 fragment from a row-major LDS tile (uint view)
__device__ __forceinline__ void frag_row(const unsigned int* tile, int row_base,
                                         int lane, FragBF& f) {
  const int half = lane >> 4;
  const unsigned int* p = tile + (size_t)(row_base + (lane & 15)) * ASTR_U;
  #pragma unroll
  for (int v = 0; v < 8; ++v) f.d[v] = p[kmap(v, half) >> 1];
}

// B 32x16 fragment ("column-major" load) from a k-major LDS tile
__device__ __forceinline__ void frag_col(const unsigned short* tile, int n_base,
                                         int lane, FragBF& f) {
  const int half = lane >> 4;
  const int n = n_base + (lane & 15);
  #pragma unroll
  for (int v = 0; v < 8; ++v) {
    const int k = kmap(v, half);
    unsigned int lo = tile[k * BSTR_H + n];
    unsigned int hi = tile[(k + 1) * BSTR_H + n];
    f.d[v] = lo | (hi << 16);
  }
}

__device__ __forceinline__ v8f wmma_bf16(const FragBF& a, const FragBF& b, v8f c) {
  return __builtin_amdgcn_wmma_f32_16x16x32_bf16(false, a.v, false, b.v,
                                                 (short)0, c, false, false);
}

// 128x32 bf16 tile copy, 512 16B chunks over 256 threads
__device__ __forceinline__ void copy_tile_A(unsigned short* dst,
                                            const unsigned short* src,
                                            int ld, int tid) {
  #pragma unroll
  for (int i = 0; i < 2; ++i) {
    const int c = tid + i * 256;
    const int row = c >> 2;
    const int q = (c & 3) * 8;
    cp16(dst + row * ASTR_H + q, src + (size_t)row * ld + q);
  }
}

// 32x128 bf16 tile copy (k-major), 512 16B chunks over 256 threads
__device__ __forceinline__ void copy_tile_B(unsigned short* dst,
                                            const unsigned short* src,
                                            int ld, int tid) {
  #pragma unroll
  for (int i = 0; i < 2; ++i) {
    const int c = tid + i * 256;
    const int row = c >> 4;
    const int q = (c & 15) * 8;
    cp16(dst + row * BSTR_H + q, src + (size_t)row * ld + q);
  }
}

// ---------------- fp32 -> bf16 conversion (4 elems/thread) ----------------
__global__ __launch_bounds__(256)
void k_cvt4(const float4* __restrict__ in, uint2* __restrict__ out, int n4) {
  int i = blockIdx.x * 256 + threadIdx.x;
  if (i < n4) {
    float4 f = in[i];
    uint2 o;
    o.x = (unsigned int)f32_to_bf16(f.x) | ((unsigned int)f32_to_bf16(f.y) << 16);
    o.y = (unsigned int)f32_to_bf16(f.z) | ((unsigned int)f32_to_bf16(f.w) << 16);
    out[i] = o;
  }
}

// ---------------- generic bf16 GEMM: C[M,N] = A[M,K] * B[K,N] ----------------
// A row-major (lda==K), B row-major (ldb==N), bf16 out. Per-z strides for batching.
// causal!=0: K-loop truncated to m0+128 (valid for P*V since P[t,s]=0 for s>t).
__global__ __launch_bounds__(256)
void k_gemm_bf16(const unsigned short* __restrict__ A,
                 const unsigned short* __restrict__ Bm,
                 unsigned short* __restrict__ C,
                 int M, int N, int K,
                 unsigned long long strideA, unsigned long long strideB,
                 unsigned long long strideC, int causal) {
  __shared__ __align__(16) unsigned short As_h[2][ATILE_H];
  __shared__ __align__(16) unsigned short Bs_h[2][BTILE_H];
  const int tid  = threadIdx.x;
  const int lane = tid & 31;
  const int wave = tid >> 5;
  const int wm = wave & 1;       // 2 waves in M
  const int wn = wave >> 1;      // 4 waves in N
  const int m0 = blockIdx.y * 128;
  const int n0 = blockIdx.x * 128;
  const unsigned short* Ap = A  + (unsigned long long)blockIdx.z * strideA;
  const unsigned short* Bp = Bm + (unsigned long long)blockIdx.z * strideB;
  unsigned short*       Cp = C  + (unsigned long long)blockIdx.z * strideC;
  const unsigned short* Atile = Ap + (size_t)m0 * K;   // + k0
  const unsigned short* Btile = Bp + n0;               // + k0*N

  const v8f zero = {0.f, 0.f, 0.f, 0.f, 0.f, 0.f, 0.f, 0.f};
  v8f acc[4][2];
  #pragma unroll
  for (int i = 0; i < 4; ++i)
    #pragma unroll
    for (int j = 0; j < 2; ++j) acc[i][j] = zero;

  const int kmax = causal ? (m0 + 128) : K;

  copy_tile_A(As_h[0], Atile, K, tid);
  copy_tile_B(Bs_h[0], Btile, N, tid);
  wait_async();
  __syncthreads();

  for (int k0 = 0; k0 < kmax; k0 += 32) {
    const int cur = (k0 >> 5) & 1;
    if (k0 + 32 < kmax) {   // prefetch next tiles into the other buffer
      copy_tile_A(As_h[cur ^ 1], Atile + (k0 + 32), K, tid);
      copy_tile_B(Bs_h[cur ^ 1], Btile + (size_t)(k0 + 32) * N, N, tid);
    }
    FragBF af[4], bfrag[2];
    #pragma unroll
    for (int mt = 0; mt < 4; ++mt)
      frag_row((const unsigned int*)As_h[cur], wm * 64 + mt * 16, lane, af[mt]);
    #pragma unroll
    for (int nt = 0; nt < 2; ++nt)
      frag_col(Bs_h[cur], wn * 32 + nt * 16, lane, bfrag[nt]);
    #pragma unroll
    for (int mt = 0; mt < 4; ++mt)
      #pragma unroll
      for (int nt = 0; nt < 2; ++nt)
        acc[mt][nt] = wmma_bf16(af[mt], bfrag[nt], acc[mt][nt]);
    wait_async();
    __syncthreads();
  }

  // C layout: lane n = lane&15; VGPR j -> M = base + j (+8 for lanes 16-31)
  const int half = lane >> 4;
  #pragma unroll
  for (int mt = 0; mt < 4; ++mt) {
    #pragma unroll
    for (int nt = 0; nt < 2; ++nt) {
      const int mb = m0 + wm * 64 + mt * 16 + half * 8;
      const int nb = n0 + wn * 32 + nt * 16 + (lane & 15);
      #pragma unroll
      for (int j = 0; j < 8; ++j)
        Cp[(size_t)(mb + j) * N + nb] = f32_to_bf16(acc[mt][nt][j]);
    }
  }
}

// ---------------- scores: S = scale * Q * K^T  (per (h,b)) ----------------
// WMMA's B operand is loaded column-major and columns of K^T are rows of K,
// so BOTH operands stage as coalesced row tiles. Upper-triangular tiles skipped.
__global__ __launch_bounds__(256)
void k_scores(const unsigned short* __restrict__ Q,
              const unsigned short* __restrict__ Kmat,
              unsigned short* __restrict__ S, float scale) {
  if (blockIdx.x > blockIdx.y) return;   // tile fully above causal diagonal
  __shared__ __align__(16) unsigned short Qs_h[2][ATILE_H];
  __shared__ __align__(16) unsigned short Ks_h[2][ATILE_H];
  const int tid  = threadIdx.x;
  const int lane = tid & 31;
  const int wave = tid >> 5;
  const int wm = wave & 1, wn = wave >> 1;
  const int m0 = blockIdx.y * 128;   // query rows
  const int n0 = blockIdx.x * 128;   // key rows
  const unsigned long long z = blockIdx.z;
  const unsigned short* Qp = Q    + z * (unsigned long long)(Tt * Dd) + (size_t)m0 * Dd;
  const unsigned short* Kp = Kmat + z * (unsigned long long)(Tt * Dd) + (size_t)n0 * Dd;
  unsigned short*       Sp = S    + z * (unsigned long long)(Tt * Tt);

  const v8f zero = {0.f, 0.f, 0.f, 0.f, 0.f, 0.f, 0.f, 0.f};
  v8f acc[4][2];
  #pragma unroll
  for (int i = 0; i < 4; ++i)
    #pragma unroll
    for (int j = 0; j < 2; ++j) acc[i][j] = zero;

  copy_tile_A(Qs_h[0], Qp, Dd, tid);
  copy_tile_A(Ks_h[0], Kp, Dd, tid);
  wait_async();
  __syncthreads();

  for (int k0 = 0; k0 < Dd; k0 += 32) {
    const int cur = (k0 >> 5) & 1;
    if (k0 + 32 < Dd) {
      copy_tile_A(Qs_h[cur ^ 1], Qp + (k0 + 32), Dd, tid);
      copy_tile_A(Ks_h[cur ^ 1], Kp + (k0 + 32), Dd, tid);
    }
    FragBF af[4], bfrag[2];
    #pragma unroll
    for (int mt = 0; mt < 4; ++mt)
      frag_row((const unsigned int*)Qs_h[cur], wm * 64 + mt * 16, lane, af[mt]);
    #pragma unroll
    for (int nt = 0; nt < 2; ++nt)
      frag_row((const unsigned int*)Ks_h[cur], wn * 32 + nt * 16, lane, bfrag[nt]);
    #pragma unroll
    for (int mt = 0; mt < 4; ++mt)
      #pragma unroll
      for (int nt = 0; nt < 2; ++nt)
        acc[mt][nt] = wmma_bf16(af[mt], bfrag[nt], acc[mt][nt]);
    wait_async();
    __syncthreads();
  }

  const int half = lane >> 4;
  #pragma unroll
  for (int mt = 0; mt < 4; ++mt) {
    #pragma unroll
    for (int nt = 0; nt < 2; ++nt) {
      const int mb = m0 + wm * 64 + mt * 16 + half * 8;
      const int nb = n0 + wn * 32 + nt * 16 + (lane & 15);
      #pragma unroll
      for (int j = 0; j < 8; ++j)
        Sp[(size_t)(mb + j) * Tt + nb] = f32_to_bf16(acc[mt][nt][j] * scale);
    }
  }
}

// ---------------- causal row softmax over S (in place, bf16) ----------------
// Writes zeros for s>t so P*V is a plain GEMM afterwards.
__global__ __launch_bounds__(256)
void k_softmax(unsigned short* __restrict__ S) {
  const int t = blockIdx.x;
  const unsigned long long z = blockIdx.y;
  unsigned short* row = S + (z * (unsigned long long)Tt + t) * (unsigned long long)Tt;
  __shared__ float red[256];
  const int tid = threadIdx.x;

  float mx = -3.0e38f;
  for (int s = tid; s <= t; s += 256) mx = fmaxf(mx, bf16_to_f32(row[s]));
  red[tid] = mx; __syncthreads();
  #pragma unroll
  for (int o = 128; o > 0; o >>= 1) {
    if (tid < o) red[tid] = fmaxf(red[tid], red[tid + o]);
    __syncthreads();
  }
  mx = red[0]; __syncthreads();

  float sum = 0.f;
  for (int s = tid; s <= t; s += 256) sum += __expf(bf16_to_f32(row[s]) - mx);
  red[tid] = sum; __syncthreads();
  #pragma unroll
  for (int o = 128; o > 0; o >>= 1) {
    if (tid < o) red[tid] += red[tid + o];
    __syncthreads();
  }
  const float inv = 1.0f / red[0];
  __syncthreads();

  for (int s = tid; s < Tt; s += 256) {
    float p = (s <= t) ? __expf(bf16_to_f32(row[s]) - mx) * inv : 0.0f;
    row[s] = f32_to_bf16(p);
  }
}

// ---------------- output projection: out = cat(AO) @ Wp + bp (f32 out) ----------------
// Head-concat permutation folded into A-row addressing: global row r = b*(H*T)+h*T+t
// reads AO[h][b][t][:]; 128-row tiles never straddle (b,h) boundaries.
__global__ __launch_bounds__(256)
void k_proj(const unsigned short* __restrict__ AO,
            const unsigned short* __restrict__ Wp,
            const float* __restrict__ bias,
            float* __restrict__ Out) {
  __shared__ __align__(16) unsigned short As_h[2][ATILE_H];
  __shared__ __align__(16) unsigned short Bs_h[2][BTILE_H];
  const int tid  = threadIdx.x;
  const int lane = tid & 31;
  const int wave = tid >> 5;
  const int wm = wave & 1, wn = wave >> 1;
  const int m0 = blockIdx.y * 128;
  const int n0 = blockIdx.x * 128;
  const int b   = m0 / (Hh * Tt);
  const int rem = m0 % (Hh * Tt);
  const int h   = rem / Tt;
  const int t0  = rem % Tt;
  const unsigned short* Ap = AO + (((size_t)h * Bb + b) * Tt + t0) * (size_t)Dd;
  const unsigned short* Btile = Wp + n0;

  const v8f zero = {0.f, 0.f, 0.f, 0.f, 0.f, 0.f, 0.f, 0.f};
  v8f acc[4][2];
  #pragma unroll
  for (int i = 0; i < 4; ++i)
    #pragma unroll
    for (int j = 0; j < 2; ++j) acc[i][j] = zero;

  copy_tile_A(As_h[0], Ap, Dd, tid);
  copy_tile_B(Bs_h[0], Btile, Dd, tid);
  wait_async();
  __syncthreads();

  for (int k0 = 0; k0 < Ee; k0 += 32) {
    const int cur = (k0 >> 5) & 1;
    if (k0 + 32 < Ee) {
      copy_tile_A(As_h[cur ^ 1], Ap + (k0 + 32), Dd, tid);
      copy_tile_B(Bs_h[cur ^ 1], Btile + (size_t)(k0 + 32) * Dd, Dd, tid);
    }
    FragBF af[4], bfrag[2];
    #pragma unroll
    for (int mt = 0; mt < 4; ++mt)
      frag_row((const unsigned int*)As_h[cur], wm * 64 + mt * 16, lane, af[mt]);
    #pragma unroll
    for (int nt = 0; nt < 2; ++nt)
      frag_col(Bs_h[cur], wn * 32 + nt * 16, lane, bfrag[nt]);
    #pragma unroll
    for (int mt = 0; mt < 4; ++mt)
      #pragma unroll
      for (int nt = 0; nt < 2; ++nt)
        acc[mt][nt] = wmma_bf16(af[mt], bfrag[nt], acc[mt][nt]);
    wait_async();
    __syncthreads();
  }

  const int half = lane >> 4;
  #pragma unroll
  for (int mt = 0; mt < 4; ++mt) {
    #pragma unroll
    for (int nt = 0; nt < 2; ++nt) {
      const int mb = m0 + wm * 64 + mt * 16 + half * 8;
      const int nb = n0 + wn * 32 + nt * 16 + (lane & 15);
      const float bv = bias[nb];
      #pragma unroll
      for (int j = 0; j < 8; ++j)
        Out[(size_t)(mb + j) * Dd + nb] = acc[mt][nt][j] + bv;
    }
  }
}

// ---------------- host launch ----------------
extern "C" void kernel_launch(void* const* d_in, const int* in_sizes, int n_in,
                              void* d_out, int out_size, void* d_ws, size_t ws_size,
                              hipStream_t stream) {
  (void)in_sizes; (void)n_in; (void)out_size; (void)ws_size;
  const float* x  = (const float*)d_in[0];
  const float* Wq = (const float*)d_in[1];
  const float* Wk = (const float*)d_in[2];
  const float* Wv = (const float*)d_in[3];
  const float* Wp = (const float*)d_in[4];
  const float* bp = (const float*)d_in[5];
  float* out = (float*)d_out;

  char* ws = (char*)d_ws;
  unsigned short* xb  = (unsigned short*)(ws + 0ull);          //  3,145,728 el
  unsigned short* wqb = (unsigned short*)(ws + 6291456ull);    //  7,077,888 el
  unsigned short* wkb = (unsigned short*)(ws + 20447232ull);
  unsigned short* wvb = (unsigned short*)(ws + 34603008ull);
  unsigned short* wpb = (unsigned short*)(ws + 48758784ull);   //    589,824 el
  unsigned short* Qb  = (unsigned short*)(ws + 49938432ull);   // 37,748,736 el
  unsigned short* Kb  = (unsigned short*)(ws + 125435904ull);
  unsigned short* Vb  = (unsigned short*)(ws + 200933376ull);
  unsigned short* Sb  = (unsigned short*)(ws + 276430848ull);  // 50,331,648 el
  unsigned short* AOb = (unsigned short*)(ws + 377094144ull);  // 37,748,736 el

  const dim3 blk(256);

  // fp32 -> bf16 conversions (float4 in, 4x bf16 out)
  k_cvt4<<<dim3((786432 + 255) / 256),  blk, 0, stream>>>((const float4*)x,  (uint2*)xb,  786432);
  k_cvt4<<<dim3((1769472 + 255) / 256), blk, 0, stream>>>((const float4*)Wq, (uint2*)wqb, 1769472);
  k_cvt4<<<dim3((1769472 + 255) / 256), blk, 0, stream>>>((const float4*)Wk, (uint2*)wkb, 1769472);
  k_cvt4<<<dim3((1769472 + 255) / 256), blk, 0, stream>>>((const float4*)Wv, (uint2*)wvb, 1769472);
  k_cvt4<<<dim3((147456 + 255) / 256),  blk, 0, stream>>>((const float4*)Wp, (uint2*)wpb, 147456);

  // QKV projections: per head, [4096x768] x [768x768] -> [H][B*T][768] bf16
  const dim3 gqkv(6, 32, 12);
  k_gemm_bf16<<<gqkv, blk, 0, stream>>>(xb, wqb, Qb, 4096, 768, 768,
                                        0ull, 589824ull, 3145728ull, 0);
  k_gemm_bf16<<<gqkv, blk, 0, stream>>>(xb, wkb, Kb, 4096, 768, 768,
                                        0ull, 589824ull, 3145728ull, 0);
  k_gemm_bf16<<<gqkv, blk, 0, stream>>>(xb, wvb, Vb, 4096, 768, 768,
                                        0ull, 589824ull, 3145728ull, 0);

  // scores (lower-triangular tiles only), scale = D^-0.5
  const float scale = 1.0f / sqrtf(768.0f);
  k_scores<<<dim3(8, 8, 48), blk, 0, stream>>>(Qb, Kb, Sb, scale);

  // causal softmax, in place on S -> P
  k_softmax<<<dim3(1024, 48), blk, 0, stream>>>(Sb);

  // head_out = P @ V, causal K-loop truncation
  k_gemm_bf16<<<dim3(6, 8, 48), blk, 0, stream>>>(Sb, Vb, AOb, 1024, 768, 1024,
                                                  1048576ull, 786432ull, 786432ull, 1);

  // out = cat @ Wp + bp (fp32)
  k_proj<<<dim3(6, 384), blk, 0, stream>>>(AOb, wpb, bp, out);
}
